// JasFourBody_45749991637049
// MI455X (gfx1250) — matrix operand order repaired and checked by reading, not compiled, not run on hardware
//
#include <hip/hip_runtime.h>

// ---------------------------------------------------------------------------
// JasFourBody on MI455X (gfx1250):
//   out[n] = sum_{i<j<k<l} kernel[m(i,j,k,l)] * x_i x_j x_k x_l
// Reformulated as a 4096x4096x4096 f16 WMMA GEMM  Q = P * W^T  with a fused
// quadratic-form epilogue. W, P (f16) total 64 MiB -> L2-resident (192 MB).
// This round: double-buffered LDS + register-staged prefetch (1 barrier per
// K-step), closed-form enumeration of non-zero simplex K-steps (~29% of dense
// work), and all 6 fragments hoisted so the 8 WMMAs issue back-to-back.
// ---------------------------------------------------------------------------

typedef __attribute__((ext_vector_type(16))) _Float16 v16h;
typedef __attribute__((ext_vector_type(8)))  float    v8f;
typedef __attribute__((ext_vector_type(4)))  float    vf4;

#define N_SITES 64
#define TN      4096      // N_SITES^2 (GEMM K and N dimension)
#define BM      128       // sample-block tile
#define BN      128       // ij-block tile
#define BK      32        // K step (matches 16x16x32 WMMA)
#define LDT     40        // padded LDS row stride in halves (80 B, 16B-aligned)

union Frag { vf4 q[2]; v16h h; };

// --- kernel 1: scatter compressed simplex weights into dense W (f16) -------
__global__ __launch_bounds__(256) void scatter_w(
    const float* __restrict__ kern,
    const int* __restrict__ ii, const int* __restrict__ jj,
    const int* __restrict__ kk, const int* __restrict__ ll,
    _Float16* __restrict__ W, int M) {
  int m = blockIdx.x * 256 + threadIdx.x;
  if (m < M) {
    int row = ii[m] * N_SITES + jj[m];
    int col = kk[m] * N_SITES + ll[m];
    W[(size_t)row * TN + col] = (_Float16)kern[m];
  }
}

// --- kernel 2: pair products P[n, k*64+l] = x[n,k]*x[n,l] (f16) ------------
__global__ __launch_bounds__(256) void pair_p(const float* __restrict__ x,
                                              _Float16* __restrict__ P) {
  __shared__ float xs[N_SITES];
  int n = blockIdx.x;
  if (threadIdx.x < N_SITES) xs[threadIdx.x] = x[(size_t)n * N_SITES + threadIdx.x];
  __syncthreads();
  _Float16* row = P + (size_t)n * TN;
  for (int e = threadIdx.x; e < TN; e += 256) {
    row[e] = (_Float16)(xs[e >> 6] * xs[e & 63]);
  }
}

// --- kernel 3: WMMA GEMM Q = P * W^T with fused x_i x_j quadratic epilogue --
// grid = (TN/BN, NS/BM), 256 threads = 8 waves in a 2x4 (row x col) layout.
// Each wave owns a 64x32 slab = 4x2 WMMA 16x16 tiles.
__global__ __launch_bounds__(256) void gemm_fused(
    const _Float16* __restrict__ P, const _Float16* __restrict__ W,
    const float* __restrict__ x, float* __restrict__ part, int NS) {
  __shared__ _Float16 sA[2][BM * LDT];  // 2 x 10240 B
  __shared__ _Float16 sB[2][BN * LDT];  // 2 x 10240 B
  __shared__ float    sRed[BM][4];      //     2048 B

  const int bn   = blockIdx.x;          // ij block
  const int bm   = blockIdx.y;          // sample block
  const int tid  = threadIdx.x;
  const int lane = tid & 31;
  const int wave = tid >> 5;            // 0..7
  const int wr   = wave >> 2;           // 0..1  (M direction)
  const int wc   = wave & 3;            // 0..3  (N direction)
  const int lc   = lane & 15;           // column / row-in-tile lane
  const int lh   = lane >> 4;           // lane half-group (K split)

  // ---- closed-form enumeration of non-zero K-steps (i<j<k<l simplex) ----
  // K-step ks covers k = ks>>1, l in [(ks&1)*32, +32). With i in {2bn,2bn+1}:
  // pairs (k, h=0|1) for k in [kmin, 30], singles (k, h=1) for k in [s0, 62].
  const int kmin   = 2 * bn + 2;
  const int npair  = (kmin <= 30) ? (31 - kmin) : 0;
  const int s0     = (kmin > 31) ? kmin : 31;
  const int nsing  = (s0 <= 62) ? (63 - s0) : 0;
  const int nsteps = 2 * npair + nsing;

  auto ks_of = [&](int s) -> int {
    if (s < 2 * npair) return 2 * (kmin + (s >> 1)) + (s & 1);
    return 2 * (s0 + (s - 2 * npair)) + 1;
  };

  v8f acc[4][2] = {};

  // global->LDS staging: each thread owns one (row, 16-half segment)
  const int ldRow = tid >> 1;
  const int ldSeg = tid & 1;
  const _Float16* gA = P + (size_t)(bm * BM + ldRow) * TN + ldSeg * 16;
  const _Float16* gB = W + (size_t)(bn * BN + ldRow) * TN + ldSeg * 16;
  const int lof = ldRow * LDT + ldSeg * 16;

  // ---- prologue: stage step 0 ----
  if (nsteps > 0) {
    const int kh = ks_of(0) * BK;
    vf4 a0 = *(const vf4*)(gA + kh), a1 = *(const vf4*)(gA + kh + 8);
    vf4 b0 = *(const vf4*)(gB + kh), b1 = *(const vf4*)(gB + kh + 8);
    ((vf4*)&sA[0][lof])[0] = a0; ((vf4*)&sA[0][lof])[1] = a1;
    ((vf4*)&sB[0][lof])[0] = b0; ((vf4*)&sB[0][lof])[1] = b1;
  }
  __syncthreads();

  // ---- software-pipelined main loop: one barrier per K-step ----
  for (int s = 0; s < nsteps; ++s) {
    const int cur = s & 1;

    // register-staged prefetch of next K-step (overlaps with WMMAs below)
    vf4 a0, a1, b0, b1;
    const bool pf = (s + 1 < nsteps);
    if (pf) {
      const int kh = ks_of(s + 1) * BK;
      a0 = *(const vf4*)(gA + kh); a1 = *(const vf4*)(gA + kh + 8);
      b0 = *(const vf4*)(gB + kh); b1 = *(const vf4*)(gB + kh + 8);
    }

    // hoist ALL fragments into distinct registers so the 8 WMMAs can issue
    // back-to-back without intervening s_wait_dscnt full stops.
    Frag af[4], bf[2];
#pragma unroll
    for (int nt = 0; nt < 2; ++nt) {
      // B fragment: lane holds B column N=lc, K = lh*16 .. +16 contiguous
      const _Float16* pb = &sB[cur][(wc * 32 + nt * 16 + lc) * LDT + lh * 16];
      bf[nt].q[0] = ((const vf4*)pb)[0];
      bf[nt].q[1] = ((const vf4*)pb)[1];
    }
#pragma unroll
    for (int mt = 0; mt < 4; ++mt) {
      // A fragment: lane holds row M=lc, K = lh*16 .. +16 contiguous
      const _Float16* pa = &sA[cur][(wr * 64 + mt * 16 + lc) * LDT];
      af[mt].q[0] = *(const vf4*)(pa + lh * 16);
      af[mt].q[1] = *(const vf4*)(pa + lh * 16 + 8);
    }

    // stage next tile into the other buffer (its readers all passed the
    // barrier that ended step s-1, so this is race-free pre-barrier)
    if (pf) {
      ((vf4*)&sA[cur ^ 1][lof])[0] = a0; ((vf4*)&sA[cur ^ 1][lof])[1] = a1;
      ((vf4*)&sB[cur ^ 1][lof])[0] = b0; ((vf4*)&sB[cur ^ 1][lof])[1] = b1;
    }

#pragma unroll
    for (int mt = 0; mt < 4; ++mt) {
#pragma unroll
      for (int nt = 0; nt < 2; ++nt) {
        acc[mt][nt] = __builtin_amdgcn_wmma_f32_16x16x32_f16(
            /*neg_a=*/false, af[mt].h, /*neg_b=*/false, bf[nt].h,
            /*c_mod=*/(short)0, acc[mt][nt],
            /*reuse_a=*/false, /*reuse_b=*/false);
      }
    }
    __syncthreads();
  }

  // Fused epilogue: out-partial[n] += sum_ij Q[n,ij] * x[n,i] * x[n,j].
  // C/D layout: VGPR r of lane L holds row M = 8*(L/16)+r, col N = L%16.
#pragma unroll
  for (int mt = 0; mt < 4; ++mt) {
#pragma unroll
    for (int r = 0; r < 8; ++r) {
      const int lrow = wr * 64 + mt * 16 + lh * 8 + r;
      const int n    = bm * BM + lrow;
      const float* xr = x + (size_t)n * N_SITES;
      float s = 0.f;
#pragma unroll
      for (int nt = 0; nt < 2; ++nt) {
        const int ij = bn * BN + wc * 32 + nt * 16 + lc;
        s += acc[mt][nt][r] * xr[ij >> 6] * xr[ij & 63];
      }
      // reduce across the 16 column lanes of this half-wave
      s += __shfl_xor(s, 1);
      s += __shfl_xor(s, 2);
      s += __shfl_xor(s, 4);
      s += __shfl_xor(s, 8);
      if (lc == 0) sRed[lrow][wc] = s;   // unique (row, wavecol) slot
    }
  }
  __syncthreads();
  if (tid < BM) {
    float v = sRed[tid][0] + sRed[tid][1] + sRed[tid][2] + sRed[tid][3];
    part[(size_t)bn * NS + (size_t)bm * BM + tid] = v;
  }
}

// --- kernel 4: deterministic final reduction over the 32 ij-blocks ----------
__global__ __launch_bounds__(256) void reduce_out(const float* __restrict__ part,
                                                  float* __restrict__ out, int NS) {
  int n = blockIdx.x * 256 + threadIdx.x;
  if (n < NS) {
    float s = 0.f;
#pragma unroll
    for (int b = 0; b < TN / BN; ++b) s += part[(size_t)b * NS + n];
    out[n] = s;
  }
}

// ---------------------------------------------------------------------------
extern "C" void kernel_launch(void* const* d_in, const int* in_sizes, int n_in,
                              void* d_out, int out_size, void* d_ws, size_t ws_size,
                              hipStream_t stream) {
  const float* x    = (const float*)d_in[0];
  const float* kern = (const float*)d_in[1];
  const int*   ii   = (const int*)d_in[2];
  const int*   jj   = (const int*)d_in[3];
  const int*   kk   = (const int*)d_in[4];
  const int*   ll   = (const int*)d_in[5];
  const int M  = in_sizes[1];
  const int NS = in_sizes[0] / N_SITES;          // 4096

  // workspace layout: W (32 MiB) | P (32 MiB) | part (0.5 MiB)
  char* ws = (char*)d_ws;
  _Float16* W    = (_Float16*)ws;
  _Float16* P    = (_Float16*)(ws + (size_t)TN * TN * sizeof(_Float16));
  float*    part = (float*)(ws + (size_t)TN * TN * sizeof(_Float16)
                               + (size_t)NS * TN * sizeof(_Float16));
  float* out = (float*)d_out;

  hipMemsetAsync(W, 0, (size_t)TN * TN * sizeof(_Float16), stream);
  scatter_w<<<(M + 255) / 256, 256, 0, stream>>>(kern, ii, jj, kk, ll, W, M);
  pair_p<<<NS, 256, 0, stream>>>(x, P);
  dim3 grid(TN / BN, NS / BM);
  gemm_fused<<<grid, 256, 0, stream>>>(P, W, x, part, NS);
  reduce_out<<<(NS + 255) / 256, 256, 0, stream>>>(part, out, NS);
}
